// SpatialTemporalGNN_36971078484249
// MI455X (gfx1250) — matrix operand order repaired and checked by reading, not compiled, not run on hardware
//
#include <hip/hip_runtime.h>
#include <hip/hip_bf16.h>

typedef _Float16 v16h __attribute__((ext_vector_type(16)));
typedef _Float16 v8h  __attribute__((ext_vector_type(8)));
typedef _Float16 v2h  __attribute__((ext_vector_type(2)));
typedef float    v8f  __attribute__((ext_vector_type(8)));

constexpr int kEmbed    = 8;
constexpr int kT        = 32;
constexpr int kGraphs   = 4096;
constexpr int kN        = 102400;   // 4096 graphs * 25 nodes
constexpr int kE        = 409600;
constexpr int kLinIn    = 400;
constexpr int kLinInPad = 416;      // 13 * 32
constexpr int kLinHid   = 800;
constexpr int kKPad     = 96;       // 32 (x chain, K=24 padded) + 64 (h chain, K=48 padded)
constexpr int kTiles    = kN / 16;  // 6400
constexpr int kLDH      = 104;      // LDS row stride in halves: 52 dwords; 52*lr mod 64
                                    // covers all 64 banks across 16 lanes -> conflict-free b128

__device__ __forceinline__ float sigmoidf_(float x) { return 1.0f / (1.0f + expf(-x)); }

// ---------------- small utility kernels ----------------

__global__ void zero_f32_kernel(float* __restrict__ p, int n) {
  int i = blockIdx.x * blockDim.x + threadIdx.x;
  if (i < n) p[i] = 0.0f;
}

__global__ void deg_kernel(const int* __restrict__ src, const float* __restrict__ w,
                           float* __restrict__ deg) {
  int e = blockIdx.x * blockDim.x + threadIdx.x;
  if (e < kE) atomicAdd(deg + src[e], w[e]);
}

__global__ void norm_kernel(const int* __restrict__ src, const int* __restrict__ dst,
                            const float* __restrict__ w, const float* __restrict__ deg,
                            float* __restrict__ norm) {
  int e = blockIdx.x * blockDim.x + threadIdx.x;
  if (e >= kE) return;
  float ds = deg[src[e]], dd = deg[dst[e]];
  float is = ds > 0.0f ? rsqrtf(ds) : 0.0f;
  float id = dd > 0.0f ? rsqrtf(dd) : 0.0f;
  norm[e] = -is * w[e] * id;
}

// out[dst] += norm * in[src], one thread per (edge, channel)
template <int C>
__global__ void spmv_kernel(const float* __restrict__ in, float* __restrict__ out,
                            const int* __restrict__ src, const int* __restrict__ dst,
                            const float* __restrict__ norm) {
  int i = blockIdx.x * blockDim.x + threadIdx.x;
  if (i >= kE * C) return;
  int e = i / C, c = i % C;
  float v = norm[e] * in[src[e] * C + c];
  atomicAdd(out + dst[e] * C + c, v);
}

// Build effective stacked weights: out = Tx0@(W0-W2) + Tx1@W1 + S@(2*W2), f16 [96][16]
__global__ void prep_gate_kernel(const float* __restrict__ Wx,  // [3][8][16]
                                 const float* __restrict__ Wh,  // [3][16][16]
                                 const float* __restrict__ bx, const float* __restrict__ bh,
                                 _Float16* __restrict__ Bo, float* __restrict__ biasO) {
  int i = blockIdx.x * blockDim.x + threadIdx.x;
  if (i < 16) biasO[i] = bx[i] + bh[i];
  if (i >= kKPad * 16) return;
  int r = i / 16, oc = i % 16;
  float v = 0.0f;
  if (r < 32) {
    if (r < 8)       v = Wx[r * 16 + oc] - Wx[(16 + r) * 16 + oc];
    else if (r < 16) v = Wx[(8 + (r - 8)) * 16 + oc];
    else if (r < 24) v = 2.0f * Wx[(16 + (r - 16)) * 16 + oc];
  } else {
    int rr = r - 32;
    if (rr < 16)      v = Wh[rr * 16 + oc] - Wh[(32 + rr) * 16 + oc];
    else if (rr < 32) v = Wh[(16 + (rr - 16)) * 16 + oc];
    else if (rr < 48) v = 2.0f * Wh[(32 + (rr - 32)) * 16 + oc];
  }
  Bo[i] = (_Float16)v;
}

// ---------------- WMMA tile helpers ----------------
// A-operand swizzle (V_WMMA_F32_16X16X32_F16): lane holds row M=lane&15,
// half h maps to k = 32*b + (h>>3)*16 + (lane>=16)*8 + (h&7)  ->  two contiguous
// 8-half chunks at k0 = 32b + 8*hi and k0+16: two ds_load_b128 per operand.
// B stored row-major [K][16]: lane holds column N=lane&15, same k order as A (strided gather).
// C/D: element j of v8f = D[m = j + (lane>=16)*8][n = lane&15].

// stage one 16-row tile [16][96] as f16 into LDS (row stride kLDH halves)
__device__ __forceinline__ void stage_tile_f16(
    _Float16* __restrict__ sh, int lane, int r0,
    const float* __restrict__ a0, const float* __restrict__ a1, const float* __restrict__ a2,
    const float* __restrict__ c0, const float* __restrict__ c1, const float* __restrict__ c2) {
  // pairs never straddle source segments (all boundaries are multiples of 8)
  for (int p = lane; p < 16 * (kKPad / 2); p += 32) {
    int r = p / (kKPad / 2);
    int k = (p % (kKPad / 2)) * 2;
    int row = r0 + r;
    float v0 = 0.0f, v1 = 0.0f;
    if (k < 8)       { const float* q = a0 + row * 8 + k;        v0 = q[0]; v1 = q[1]; }
    else if (k < 16) { const float* q = a1 + row * 8 + (k - 8);  v0 = q[0]; v1 = q[1]; }
    else if (k < 24) { const float* q = a2 + row * 8 + (k - 16); v0 = q[0]; v1 = q[1]; }
    else if (k < 32) { /* zero pad */ }
    else if (k < 48) { const float* q = c0 + row * 16 + (k - 32); v0 = q[0]; v1 = q[1]; }
    else if (k < 64) { const float* q = c1 + row * 16 + (k - 48); v0 = q[0]; v1 = q[1]; }
    else if (k < 80) { const float* q = c2 + row * 16 + (k - 64); v0 = q[0]; v1 = q[1]; }
    v2h hv = { (_Float16)v0, (_Float16)v1 };
    *(v2h*)(sh + r * kLDH + k) = hv;
  }
}

__device__ __forceinline__ v16h load_a_tile(const _Float16* __restrict__ sh,
                                            int lr, int hi, int b) {
  v16h av;
  const v8h* p = (const v8h*)(sh + lr * kLDH + 32 * b + hi * 8);  // 16B aligned
  ((v8h*)&av)[0] = p[0];   // halves 0..7  -> k = 32b + 8hi + 0..7
  ((v8h*)&av)[1] = p[2];   // halves 8..15 -> k = +16
  return av;
}

__global__ void __launch_bounds__(128)
gates_zr_kernel(const float* __restrict__ x_t, const float* __restrict__ Tx1,
                const float* __restrict__ Sx, const float* __restrict__ H,
                const float* __restrict__ TH1, const float* __restrict__ Sh,
                const _Float16* __restrict__ Bz, const _Float16* __restrict__ Br,
                const float* __restrict__ biasZ, const float* __restrict__ biasR,
                float* __restrict__ Z, float* __restrict__ HR) {
  __shared__ __align__(16) _Float16 smem[4][16 * kLDH];
  const int wave = threadIdx.y, lane = threadIdx.x;
  const int tile = blockIdx.x * 4 + wave;
  const int r0 = tile * 16;
  _Float16* sh = smem[wave];

  stage_tile_f16(sh, lane, r0, x_t, Tx1, Sx, H, TH1, Sh);
  __syncthreads();

  const int lr = lane & 15;
  const int hi = (lane >> 4) & 1;
  v8f accZ = {}, accR = {};
#pragma unroll
  for (int b = 0; b < 3; ++b) {
    v16h av = load_a_tile(sh, lr, hi, b);
    v16h bz, br;
    for (int s = 0; s < 16; ++s) { int k = 32 * b + hi * 8 + (s < 8 ? s : s + 8); bz[s] = Bz[k * 16 + lr]; br[s] = Br[k * 16 + lr]; }
    accZ = __builtin_amdgcn_wmma_f32_16x16x32_f16(false, av, false, bz, (short)0, accZ, false, false);
    accR = __builtin_amdgcn_wmma_f32_16x16x32_f16(false, av, false, br, (short)0, accR, false, false);
  }
#pragma unroll
  for (int j = 0; j < 8; ++j) {
    int m = j + hi * 8;
    int idx = (r0 + m) * 16 + lr;
    float zv = sigmoidf_(accZ[j] + biasZ[lr]);
    float rv = sigmoidf_(accR[j] + biasR[lr]);
    Z[idx] = zv;
    HR[idx] = H[idx] * rv;
  }
}

__global__ void __launch_bounds__(128)
hstep_kernel(const float* __restrict__ x_t, const float* __restrict__ Tx1,
             const float* __restrict__ Sx, const float* __restrict__ HR,
             const float* __restrict__ THR1, const float* __restrict__ Shr,
             const _Float16* __restrict__ Bh, const float* __restrict__ biasH,
             const float* __restrict__ Z, float* __restrict__ H) {
  __shared__ __align__(16) _Float16 smem[4][16 * kLDH];
  const int wave = threadIdx.y, lane = threadIdx.x;
  const int tile = blockIdx.x * 4 + wave;
  const int r0 = tile * 16;
  _Float16* sh = smem[wave];

  stage_tile_f16(sh, lane, r0, x_t, Tx1, Sx, HR, THR1, Shr);
  __syncthreads();

  const int lr = lane & 15;
  const int hi = (lane >> 4) & 1;
  v8f acc = {};
#pragma unroll
  for (int b = 0; b < 3; ++b) {
    v16h av = load_a_tile(sh, lr, hi, b);
    v16h bv;
    for (int s = 0; s < 16; ++s) { int k = 32 * b + hi * 8 + (s < 8 ? s : s + 8); bv[s] = Bh[k * 16 + lr]; }
    acc = __builtin_amdgcn_wmma_f32_16x16x32_f16(false, av, false, bv, (short)0, acc, false, false);
  }
#pragma unroll
  for (int j = 0; j < 8; ++j) {
    int m = j + hi * 8;
    int idx = (r0 + m) * 16 + lr;
    float ht = tanhf(acc[j] + biasH[lr]);
    float z = Z[idx];
    H[idx] = z * H[idx] + (1.0f - z) * ht;   // in-place safe: lane reads/writes its own element
  }
}

// ---------------- classification head ----------------

__global__ void head_prep_x_kernel(const float* __restrict__ H, _Float16* __restrict__ Xh) {
  int i = blockIdx.x * blockDim.x + threadIdx.x;
  if (i >= kGraphs * kLinInPad) return;
  int g = i / kLinInPad, f = i % kLinInPad;
  float v = 0.0f;
  if (f < kLinIn) v = fmaxf(H[g * kLinIn + f], 0.0f);  // H flat is exactly [4096][400]
  Xh[i] = (_Float16)v;
}

__global__ void head_prep_w1_kernel(const float* __restrict__ W1, _Float16* __restrict__ W1h) {
  int i = blockIdx.x * blockDim.x + threadIdx.x;
  if (i >= kLinInPad * kLinHid) return;
  int k = i / kLinHid, oc = i % kLinHid;
  W1h[i] = (_Float16)(k < kLinIn ? W1[k * kLinHid + oc] : 0.0f);
}

__global__ void __launch_bounds__(128)
head_gemm1_kernel(const _Float16* __restrict__ Xh,   // [4096][416] f16
                  const _Float16* __restrict__ W1h,  // [416][800] f16
                  const float* __restrict__ b1, float* __restrict__ Y1) {  // [4096][800]
  const int lane = threadIdx.x;
  const int nt = blockIdx.x;                             // 0..49
  const int mt = blockIdx.y * 4 + threadIdx.y;           // 0..255
  const int lr = lane & 15;
  const int hi = (lane >> 4) & 1;
  const int row = mt * 16 + lr;
  const _Float16* xrow = Xh + row * kLinInPad;

  v8f acc = {};
#pragma unroll
  for (int b = 0; b < kLinInPad / 32; ++b) {             // 13 chained WMMAs, K=416
    v16h av;
    const _Float16* p = xrow + 32 * b + hi * 8;
    ((v8h*)&av)[0] = *(const v8h*)(p);
    ((v8h*)&av)[1] = *(const v8h*)(p + 16);
    v16h bv;
    for (int s = 0; s < 16; ++s) { int k = 32 * b + hi * 8 + (s < 8 ? s : s + 8); bv[s] = W1h[k * kLinHid + nt * 16 + lr]; }
    acc = __builtin_amdgcn_wmma_f32_16x16x32_f16(false, av, false, bv, (short)0, acc, false, false);
  }
#pragma unroll
  for (int j = 0; j < 8; ++j) {
    int m = mt * 16 + j + hi * 8;
    int n = nt * 16 + lr;
    Y1[m * kLinHid + n] = fmaxf(acc[j] + b1[n], 0.0f);
  }
}

__global__ void __launch_bounds__(128)
head_out_kernel(const float* __restrict__ Y1, const float* __restrict__ W2,
                const float* __restrict__ b2, float* __restrict__ out) {
  int g = blockIdx.x * 4 + threadIdx.y;
  int lane = threadIdx.x;
  const float* y = Y1 + (size_t)g * kLinHid;
  float s0 = 0.0f, s1 = 0.0f;
  for (int k = lane; k < kLinHid; k += 32) {
    float v = y[k];
    s0 += v * W2[k * 2 + 0];
    s1 += v * W2[k * 2 + 1];
  }
  for (int off = 16; off > 0; off >>= 1) {
    s0 += __shfl_xor(s0, off, 32);
    s1 += __shfl_xor(s1, off, 32);
  }
  if (lane == 0) {
    float a = s0 + b2[0], c = s1 + b2[1];
    float m = fmaxf(a, c);
    float e0 = expf(a - m), e1 = expf(c - m);
    float inv = 1.0f / (e0 + e1);
    out[g * 2 + 0] = e0 * inv;
    out[g * 2 + 1] = e1 * inv;
  }
}

// ---------------- launch ----------------

extern "C" void kernel_launch(void* const* d_in, const int* in_sizes, int n_in,
                              void* d_out, int out_size, void* d_ws, size_t ws_size,
                              hipStream_t stream) {
  (void)in_sizes; (void)n_in; (void)out_size; (void)ws_size;
  const float* x_temporal = (const float*)d_in[0];
  const int*   ei = (const int*)d_in[1];
  const float* ew = (const float*)d_in[2];
  const int* src = ei;
  const int* dst = ei + kE;
  const float* W_xz = (const float*)d_in[4];  const float* b_xz = (const float*)d_in[5];
  const float* W_hz = (const float*)d_in[6];  const float* b_hz = (const float*)d_in[7];
  const float* W_xr = (const float*)d_in[8];  const float* b_xr = (const float*)d_in[9];
  const float* W_hr = (const float*)d_in[10]; const float* b_hr = (const float*)d_in[11];
  const float* W_xh = (const float*)d_in[12]; const float* b_xh = (const float*)d_in[13];
  const float* W_hh = (const float*)d_in[14]; const float* b_hh = (const float*)d_in[15];
  const float* W1 = (const float*)d_in[16];   const float* b1 = (const float*)d_in[17];
  const float* W2 = (const float*)d_in[18];   const float* b2 = (const float*)d_in[19];
  float* out = (float*)d_out;

  // workspace layout (all 512-aligned); total ~55 MB
  char* ws = (char*)d_ws;
  size_t o = 0;
  auto alloc = [&](size_t bytes) { char* p = ws + o; o += (bytes + 511) & ~size_t(511); return p; };
  float* deg  = (float*)alloc((size_t)kN * 4);
  float* norm = (float*)alloc((size_t)kE * 4);
  float* H    = (float*)alloc((size_t)kN * 16 * 4);
  float* Z    = (float*)alloc((size_t)kN * 16 * 4);
  float* HR   = (float*)alloc((size_t)kN * 16 * 4);
  float* Tx1  = (float*)alloc((size_t)kN * 8 * 4);   // Tx1,Sx,TH1,Sh contiguous (sizes 512-mult)
  float* Sx   = (float*)alloc((size_t)kN * 8 * 4);
  float* TH1  = (float*)alloc((size_t)kN * 16 * 4);
  float* Sh   = (float*)alloc((size_t)kN * 16 * 4);
  float* THR1 = (float*)alloc((size_t)kN * 16 * 4);  // THR1,Shr contiguous
  float* Shr  = (float*)alloc((size_t)kN * 16 * 4);
  _Float16* Bz = (_Float16*)alloc(kKPad * 16 * 2);
  _Float16* Br = (_Float16*)alloc(kKPad * 16 * 2);
  _Float16* Bh = (_Float16*)alloc(kKPad * 16 * 2);
  float* biasZ = (float*)alloc(16 * 4);
  float* biasR = (float*)alloc(16 * 4);
  float* biasH = (float*)alloc(16 * 4);
  // head buffers alias recurrent scratch (dead after the time loop)
  _Float16* Xh  = (_Float16*)THR1;  // 3.4 MB <= 6.55 MB
  _Float16* W1h = (_Float16*)Shr;   // 0.67 MB
  float*    Y1  = TH1;              // 13.1 MB == TH1+Sh (contiguous)

  const int TPB = 256;
  zero_f32_kernel<<<(kN + TPB - 1) / TPB, TPB, 0, stream>>>(deg, kN);
  deg_kernel<<<(kE + TPB - 1) / TPB, TPB, 0, stream>>>(src, ew, deg);
  norm_kernel<<<(kE + TPB - 1) / TPB, TPB, 0, stream>>>(src, dst, ew, deg, norm);
  prep_gate_kernel<<<6, TPB, 0, stream>>>(W_xz, W_hz, b_xz, b_hz, Bz, biasZ);
  prep_gate_kernel<<<6, TPB, 0, stream>>>(W_xr, W_hr, b_xr, b_hr, Br, biasR);
  prep_gate_kernel<<<6, TPB, 0, stream>>>(W_xh, W_hh, b_xh, b_hh, Bh, biasH);
  zero_f32_kernel<<<(kN * 16 + TPB - 1) / TPB, TPB, 0, stream>>>(H, kN * 16);

  dim3 waveBlock(32, 4);
  for (int t = 0; t < kT; ++t) {
    const float* xt = x_temporal + (size_t)t * kN * kEmbed;
    zero_f32_kernel<<<(kN * 48 + TPB - 1) / TPB, TPB, 0, stream>>>(Tx1, kN * 48);
    spmv_kernel<8><<<(kE * 8 + TPB - 1) / TPB, TPB, 0, stream>>>(xt, Tx1, src, dst, norm);
    spmv_kernel<8><<<(kE * 8 + TPB - 1) / TPB, TPB, 0, stream>>>(Tx1, Sx, src, dst, norm);
    spmv_kernel<16><<<(kE * 16 + TPB - 1) / TPB, TPB, 0, stream>>>(H, TH1, src, dst, norm);
    spmv_kernel<16><<<(kE * 16 + TPB - 1) / TPB, TPB, 0, stream>>>(TH1, Sh, src, dst, norm);
    gates_zr_kernel<<<kTiles / 4, waveBlock, 0, stream>>>(xt, Tx1, Sx, H, TH1, Sh,
                                                          Bz, Br, biasZ, biasR, Z, HR);
    zero_f32_kernel<<<(kN * 32 + TPB - 1) / TPB, TPB, 0, stream>>>(THR1, kN * 32);
    spmv_kernel<16><<<(kE * 16 + TPB - 1) / TPB, TPB, 0, stream>>>(HR, THR1, src, dst, norm);
    spmv_kernel<16><<<(kE * 16 + TPB - 1) / TPB, TPB, 0, stream>>>(THR1, Shr, src, dst, norm);
    hstep_kernel<<<kTiles / 4, waveBlock, 0, stream>>>(xt, Tx1, Sx, HR, THR1, Shr,
                                                       Bh, biasH, Z, H);
  }

  head_prep_x_kernel<<<(kGraphs * kLinInPad + TPB - 1) / TPB, TPB, 0, stream>>>(H, Xh);
  head_prep_w1_kernel<<<(kLinInPad * kLinHid + TPB - 1) / TPB, TPB, 0, stream>>>(W1, W1h);
  dim3 g1grid(kLinHid / 16, (kGraphs / 16) / 4);  // (50, 64) tiles of 16x16
  head_gemm1_kernel<<<g1grid, waveBlock, 0, stream>>>(Xh, W1h, b1, Y1);
  head_out_kernel<<<kGraphs / 4, waveBlock, 0, stream>>>(Y1, W2, b2, out);
}
